// VariationalNet_28595892257183
// MI455X (gfx1250) — matrix operand order repaired and checked by reading, not compiled, not run on hardware
//
#include <hip/hip_runtime.h>
#include <hip/hip_bf16.h>
#include <math.h>

// ---------------------------------------------------------------------------
// Types for CDNA5 WMMA
// ---------------------------------------------------------------------------
typedef __attribute__((ext_vector_type(16))) __bf16 v16bf;
typedef __attribute__((ext_vector_type(8)))  float  v8f;

#define BM 128
#define BN 128
#define BK 32
#define LDT 40  // padded LDS row stride (halves): 80B => conflict-free b128 reads

// ---------------------------------------------------------------------------
// CDNA5 async copy: global -> LDS, 16 bytes per lane, tracked by ASYNCcnt.
// lds_byte: LDS byte offset (low 32 bits of the generic shared-pointer address,
// which per the aperture rules is the LDS offset). gaddr: 64-bit global addr.
// ---------------------------------------------------------------------------
__device__ __forceinline__ void async_copy_b128(unsigned lds_byte,
                                                unsigned long long gaddr) {
    asm volatile("global_load_async_to_lds_b128 %0, %1, off"
                 :: "v"(lds_byte), "v"(gaddr) : "memory");
}

__device__ __forceinline__ void wait_async_zero() {
    asm volatile("s_wait_asynccnt 0x0" ::: "memory");
}

// ---------------------------------------------------------------------------
// Elementwise prep: p = mu + eps * softplus(rho)
// ---------------------------------------------------------------------------
__device__ __forceinline__ float softplusf(float x) {
    return (x > 20.f) ? x : log1pf(__expf(x));
}

__global__ void prep_param_bf16(const float* __restrict__ mu,
                                const float* __restrict__ rho,
                                const float* __restrict__ eps,
                                __hip_bfloat16* __restrict__ dst, int n) {
    int i = blockIdx.x * blockDim.x + threadIdx.x;
    if (i < n) dst[i] = __float2bfloat16(mu[i] + eps[i] * softplusf(rho[i]));
}

__global__ void prep_param_f32(const float* __restrict__ mu,
                               const float* __restrict__ rho,
                               const float* __restrict__ eps,
                               float* __restrict__ dst, int n) {
    int i = blockIdx.x * blockDim.x + threadIdx.x;
    if (i < n) dst[i] = mu[i] + eps[i] * softplusf(rho[i]);
}

__global__ void cvt_f32_bf16(const float* __restrict__ src,
                             __hip_bfloat16* __restrict__ dst, int n) {
    int i = blockIdx.x * blockDim.x + threadIdx.x;
    if (i < n) dst[i] = __float2bfloat16(src[i]);
}

// ---------------------------------------------------------------------------
// C[M,N] = relu(A[M,K] @ W[N,K]^T + bias[N])   (bf16 in, bf16 out, f32 accum)
// Block: 128x128 tile, 256 threads = 8 waves (wave32), 2x4 wave grid.
// Each wave: 4 (M) x 2 (N) WMMA 16x16x32 bf16 tiles.
// K-loop double-buffered LDS fed by GLOBAL_LOAD_ASYNC_TO_LDS_B128.
// ---------------------------------------------------------------------------
__global__ __launch_bounds__(256) void gemm_bias_relu_bf16(
    const __hip_bfloat16* __restrict__ A,  // [M,K]
    const __hip_bfloat16* __restrict__ W,  // [N,K] (row-major, K contiguous)
    const float* __restrict__ bias,        // [N]
    __hip_bfloat16* __restrict__ C,        // [M,N]
    int M, int N, int K)
{
    __shared__ __align__(16) unsigned short As[2][BM * LDT];
    __shared__ __align__(16) unsigned short Bs[2][BN * LDT];

    const int tid  = threadIdx.x;
    const int lane = tid & 31;
    const int wave = tid >> 5;
    const int wm   = wave >> 2;   // 0..1  -> 64-row slab
    const int wn   = wave & 3;    // 0..3  -> 32-col slab

    const int m0 = blockIdx.y * BM;
    const int n0 = blockIdx.x * BN;

    // cooperative tile load: each thread moves 32B of A and 32B of W per k-step
    const int lrow = tid >> 1;         // 0..127
    const int lcol = (tid & 1) * 16;   // 0 or 16 (halves)

    const unsigned long long Ag =
        (unsigned long long)(const unsigned short*)A + 2ull * ((size_t)(m0 + lrow) * K + lcol);
    const unsigned long long Wg =
        (unsigned long long)(const unsigned short*)W + 2ull * ((size_t)(n0 + lrow) * K + lcol);

    // LDS byte offsets of this thread's staging slots (per double buffer)
    const unsigned ldsA0 = (unsigned)(uintptr_t)&As[0][lrow * LDT + lcol];
    const unsigned ldsA1 = (unsigned)(uintptr_t)&As[1][lrow * LDT + lcol];
    const unsigned ldsB0 = (unsigned)(uintptr_t)&Bs[0][lrow * LDT + lcol];
    const unsigned ldsB1 = (unsigned)(uintptr_t)&Bs[1][lrow * LDT + lcol];

    // fragment lane decomposition (ISA 7.12.2 layouts)
    const int fr = lane & 15;   // A: row in 16;  B: col in 16
    const int hi = lane >> 4;   // half-wave select

    v8f acc[4][2];
    const v8f vzero = {0.f, 0.f, 0.f, 0.f, 0.f, 0.f, 0.f, 0.f};
#pragma unroll
    for (int i = 0; i < 4; ++i)
#pragma unroll
        for (int j = 0; j < 2; ++j) acc[i][j] = vzero;

    const int NK = K / BK;

    // prologue: async-stage k-tile 0 into buffer 0
    async_copy_b128(ldsA0,      Ag);
    async_copy_b128(ldsA0 + 16, Ag + 16);
    async_copy_b128(ldsB0,      Wg);
    async_copy_b128(ldsB0 + 16, Wg + 16);
    wait_async_zero();
    __syncthreads();

    for (int kt = 0; kt < NK; ++kt) {
        const int cur = kt & 1;

        if (kt + 1 < NK) {  // async-stage next k-tile into the other buffer
            const unsigned long long Agn = Ag + 2ull * (size_t)(kt + 1) * BK;
            const unsigned long long Wgn = Wg + 2ull * (size_t)(kt + 1) * BK;
            const unsigned la = cur ? ldsA0 : ldsA1;
            const unsigned lb = cur ? ldsB0 : ldsB1;
            async_copy_b128(la,      Agn);
            async_copy_b128(la + 16, Agn + 16);
            async_copy_b128(lb,      Wgn);
            async_copy_b128(lb + 16, Wgn + 16);
        }

        // A fragment: lane holds row fr, halves [hi*8 .. +7] and [hi*8+16 .. +7]
        v16bf afrag[4];
#pragma unroll
        for (int i = 0; i < 4; ++i) {
            const int r = wm * 64 + i * 16 + fr;
            const uint4* p = (const uint4*)&As[cur][r * LDT + hi * 8];
            union { uint4 u[2]; v16bf v; } un;
            un.u[0] = p[0];
            un.u[1] = p[2];  // +16 halves
            afrag[i] = un.v;
        }
        // B fragment: lane holds col fr, halves [hi*8 .. +7] and [hi*8+16 .. +7]
        v16bf bfrag[2];
#pragma unroll
        for (int j = 0; j < 2; ++j) {
            const int n = wn * 32 + j * 16 + fr;
            const uint4* p = (const uint4*)&Bs[cur][n * LDT + hi * 8];
            union { uint4 u[2]; v16bf v; } un;
            un.u[0] = p[0];
            un.u[1] = p[2];
            bfrag[j] = un.v;
        }

#pragma unroll
        for (int i = 0; i < 4; ++i)
#pragma unroll
            for (int j = 0; j < 2; ++j)
                acc[i][j] = __builtin_amdgcn_wmma_f32_16x16x32_bf16(
                    false, afrag[i], false, bfrag[j], (short)0, acc[i][j],
                    false, false);

        if (kt + 1 < NK) {
            wait_async_zero();   // next tile landed in LDS
            __syncthreads();     // all waves done reading cur & see new tile
        }
    }

    // epilogue: D layout -> row = vgpr + 8*hi, col = fr ; fused bias + relu
#pragma unroll
    for (int i = 0; i < 4; ++i) {
        const int mrow = m0 + wm * 64 + i * 16 + hi * 8;
#pragma unroll
        for (int j = 0; j < 2; ++j) {
            const int ncol = n0 + wn * 32 + j * 16 + fr;
            const float bval = bias[ncol];
            __hip_bfloat16* cp = C + (size_t)mrow * N + ncol;
#pragma unroll
            for (int v = 0; v < 8; ++v) {
                float y = acc[i][j][v] + bval;
                y = y > 0.f ? y : 0.f;
                cp[(size_t)v * N] = __float2bfloat16(y);
            }
        }
    }
}

// ---------------------------------------------------------------------------
// out[m] = sum_k H[m,k]*wo[k] + bo      (one wave per row)
// ---------------------------------------------------------------------------
__global__ __launch_bounds__(256) void gemv_out(
    const __hip_bfloat16* __restrict__ H, const float* __restrict__ wo,
    const float* __restrict__ bo, float* __restrict__ out, int M, int K)
{
    const int row  = blockIdx.x * 8 + (threadIdx.x >> 5);
    const int lane = threadIdx.x & 31;
    if (row >= M) return;
    const __hip_bfloat16* h = H + (size_t)row * K;
    float s = 0.f;
    for (int k = lane; k < K; k += 32)
        s += __bfloat162float(h[k]) * wo[k];
#pragma unroll
    for (int off = 16; off > 0; off >>= 1)
        s += __shfl_down(s, off, 32);
    if (lane == 0) out[row] = s + bo[0];
}

// ---------------------------------------------------------------------------
// Launch
// ---------------------------------------------------------------------------
extern "C" void kernel_launch(void* const* d_in, const int* in_sizes, int n_in,
                              void* d_out, int out_size, void* d_ws, size_t ws_size,
                              hipStream_t stream) {
    constexpr int Bn = 4096, D = 2048, H = 3;

    const float* x       = (const float*)d_in[0];
    const float* w_mu_h  = (const float*)d_in[1];
    const float* w_rho_h = (const float*)d_in[2];
    const float* b_mu_h  = (const float*)d_in[3];
    const float* b_rho_h = (const float*)d_in[4];
    const float* w_mu_o  = (const float*)d_in[5];
    const float* w_rho_o = (const float*)d_in[6];
    const float* b_mu_o  = (const float*)d_in[7];
    const float* b_rho_o = (const float*)d_in[8];
    const float* eps_w_h = (const float*)d_in[9];
    const float* eps_b_h = (const float*)d_in[10];
    const float* eps_w_o = (const float*)d_in[11];
    const float* eps_b_o = (const float*)d_in[12];

    auto alignup = [](size_t v) { return (v + 255) & ~size_t(255); };
    char* ws = (char*)d_ws;
    size_t off = 0;

    __hip_bfloat16* Wq = (__hip_bfloat16*)(ws + off);           // 3 x D x D bf16
    off = alignup(off + (size_t)H * D * D * sizeof(__hip_bfloat16));
    float* bh = (float*)(ws + off);                             // 3 x D f32
    off = alignup(off + (size_t)H * D * sizeof(float));
    float* wo = (float*)(ws + off);                             // D f32
    off = alignup(off + (size_t)D * sizeof(float));
    float* bo = (float*)(ws + off);                             // 1 f32
    off = alignup(off + sizeof(float));
    __hip_bfloat16* h0 = (__hip_bfloat16*)(ws + off);           // Bn x D bf16
    off = alignup(off + (size_t)Bn * D * sizeof(__hip_bfloat16));
    __hip_bfloat16* h1 = (__hip_bfloat16*)(ws + off);           // Bn x D bf16
    off = alignup(off + (size_t)Bn * D * sizeof(__hip_bfloat16));
    (void)ws_size;

    // 1) materialize variational parameters
    {
        int n = H * D * D;
        prep_param_bf16<<<(n + 255) / 256, 256, 0, stream>>>(w_mu_h, w_rho_h, eps_w_h, Wq, n);
    }
    {
        int n = H * D;
        prep_param_f32<<<(n + 255) / 256, 256, 0, stream>>>(b_mu_h, b_rho_h, eps_b_h, bh, n);
    }
    prep_param_f32<<<(D + 255) / 256, 256, 0, stream>>>(w_mu_o, w_rho_o, eps_w_o, wo, D);
    prep_param_f32<<<1, 32, 0, stream>>>(b_mu_o, b_rho_o, eps_b_o, bo, 1);

    // 2) x -> bf16
    {
        int n = Bn * D;
        cvt_f32_bf16<<<(n + 255) / 256, 256, 0, stream>>>(x, h0, n);
    }

    // 3) three hidden layers (ping-pong h0/h1)
    dim3 grid(D / BN, Bn / BM);
    dim3 block(256);
    gemm_bias_relu_bf16<<<grid, block, 0, stream>>>(h0, Wq + 0 * (size_t)D * D, bh + 0 * D, h1, Bn, D, D);
    gemm_bias_relu_bf16<<<grid, block, 0, stream>>>(h1, Wq + 1 * (size_t)D * D, bh + 1 * D, h0, Bn, D, D);
    gemm_bias_relu_bf16<<<grid, block, 0, stream>>>(h0, Wq + 2 * (size_t)D * D, bh + 2 * D, h1, Bn, D, D);

    // 4) output GEMV (f32 out, out_size == 4096)
    gemv_out<<<(Bn + 7) / 8, 256, 0, stream>>>(h1, wo, bo, (float*)d_out, Bn, D);
}
